// ModernGPT2Attention_88441966559282
// MI455X (gfx1250) — matrix operand order repaired and checked by reading, not compiled, not run on hardware
//
#include <hip/hip_runtime.h>
#include <hip/hip_bf16.h>

// ---------- CDNA5 WMMA types ----------
typedef __attribute__((ext_vector_type(16))) __bf16 v16bf;
typedef __attribute__((ext_vector_type(8)))  float  v8f;
typedef __attribute__((ext_vector_type(4)))  int    v4i32;

struct BF16x16 { uint4 lo, hi; };   // 32 bytes == v16bf

// ---------- optional CDNA5 async global->LDS path ----------
#if defined(__has_builtin)
#if __has_builtin(__builtin_amdgcn_global_load_async_to_lds_b128) && \
    __has_builtin(__builtin_amdgcn_s_wait_asynccnt)
#define USE_ASYNC_LDS 1
#endif
#endif
#ifndef USE_ASYNC_LDS
#define USE_ASYNC_LDS 0
#endif

#if USE_ASYNC_LDS
typedef __attribute__((address_space(1))) v4i32* gvec_t;
typedef __attribute__((address_space(3))) v4i32* lvec_t;
__device__ inline void async_cp16(const void* g, void* l) {
  __builtin_amdgcn_global_load_async_to_lds_b128((gvec_t)g, (lvec_t)l, 0, 0);
}
__device__ inline void async_wait0() { __builtin_amdgcn_s_wait_asynccnt(0); }
#endif

__device__ inline v8f zero8() {
  v8f v;
  #pragma unroll
  for (int i = 0; i < 8; ++i) v[i] = 0.0f;
  return v;
}

__device__ inline unsigned short bfbits(float f) {
  unsigned u = __builtin_bit_cast(unsigned, f);
  unsigned r = u + 0x7FFFu + ((u >> 16) & 1u);   // round-to-nearest-even
  return (unsigned short)(r >> 16);
}
__device__ inline float bff(unsigned short h) {
  unsigned u = ((unsigned)h) << 16;
  return __builtin_bit_cast(float, u);
}
__device__ inline unsigned pack2(float lo, float hi) {
  return (unsigned)bfbits(lo) | ((unsigned)bfbits(hi) << 16);
}

// A-fragment (16x32 bf16): lanes 0-15 rows, K chunks at kofs+{0,16} (+8 for hi half-wave)
__device__ inline v16bf ld_afrag(const unsigned short* base, int stride, int kofs, int lane) {
  int r  = lane & 15;
  int k0 = kofs + ((lane >> 4) << 3);
  const unsigned short* p = base + r * stride + k0;
  BF16x16 t;
  t.lo = *(const uint4*)(p);
  t.hi = *(const uint4*)(p + 16);
  return __builtin_bit_cast(v16bf, t);
}
// B-fragment (32x16 bf16): lane = column, K chunks at kofs+{0,16} (+8 for hi half-wave)
__device__ inline v16bf ld_bfrag(const unsigned short* base, int stride, int kofs, int lane) {
  int r  = lane & 15;
  int k0 = kofs + ((lane >> 4) << 3);
  const unsigned short* p = base + r * stride + k0;
  BF16x16 t;
  t.lo = *(const uint4*)(p);
  t.hi = *(const uint4*)(p + 16);
  return __builtin_bit_cast(v16bf, t);
}

__device__ inline float rmax16(float v) {
  v = fmaxf(v, __shfl_xor(v, 1, 32));
  v = fmaxf(v, __shfl_xor(v, 2, 32));
  v = fmaxf(v, __shfl_xor(v, 4, 32));
  v = fmaxf(v, __shfl_xor(v, 8, 32));
  return v;
}
__device__ inline float rsum16(float v) {
  v += __shfl_xor(v, 1, 32);
  v += __shfl_xor(v, 2, 32);
  v += __shfl_xor(v, 4, 32);
  v += __shfl_xor(v, 8, 32);
  return v;
}

// =======================================================================
// Generic C[M,N] = A[M,K] * B[N,K]^T, bf16 WMMA, 128x128 block, 8 waves
// =======================================================================
#define BKM 128
#define BKN 128
#define BKK 32
#define LDT 40   // LDS row stride (elems); 80 bytes -> 16B aligned rows

template<bool A_BF16, bool OUT_BF16>
__global__ __launch_bounds__(256) void gemm_bf16_wmma(
    const void* __restrict__ Avp, const float* __restrict__ B,
    void* __restrict__ Cvp, int M, int N, int K)
{
  __shared__ __align__(16) unsigned short As[BKM * LDT];
  __shared__ __align__(16) unsigned short Bs[BKN * LDT];

  const int tid  = threadIdx.x;
  const int lane = tid & 31;
  const int wave = tid >> 5;              // 0..7
  const int wm   = (wave >> 2) * 64;      // waves: 2 (M) x 4 (N)
  const int wn   = (wave & 3) * 32;
  const int bm   = blockIdx.y * BKM;
  const int bn   = blockIdx.x * BKN;

  const int ldRow = tid >> 1;             // 0..127
  const int ldCol = (tid & 1) * 16;       // 0 or 16 (elems in K)

  v8f acc[4][2];
  #pragma unroll
  for (int mt = 0; mt < 4; ++mt) {
    #pragma unroll
    for (int nt = 0; nt < 2; ++nt) acc[mt][nt] = zero8();
  }

  for (int k0 = 0; k0 < K; k0 += BKK) {
    // ---- stage A tile ----
    if (A_BF16) {
      const unsigned short* Ag = (const unsigned short*)Avp;
      const unsigned short* src = Ag + (size_t)(bm + ldRow) * K + k0 + ldCol;
#if USE_ASYNC_LDS
      async_cp16(src,     &As[ldRow * LDT + ldCol]);
      async_cp16(src + 8, &As[ldRow * LDT + ldCol + 8]);
#else
      uint4* d = (uint4*)&As[ldRow * LDT + ldCol];
      d[0] = *(const uint4*)(src);
      d[1] = *(const uint4*)(src + 8);
#endif
      if (k0 + BKK < K) __builtin_prefetch(src + BKK, 0, 0);
    } else {
      const float* Ag = (const float*)Avp;
      const float4* s4 = (const float4*)(Ag + (size_t)(bm + ldRow) * K + k0 + ldCol);
      unsigned u[8];
      #pragma unroll
      for (int c = 0; c < 4; ++c) {
        float4 t = s4[c];
        u[2*c]   = pack2(t.x, t.y);
        u[2*c+1] = pack2(t.z, t.w);
      }
      uint4* d = (uint4*)&As[ldRow * LDT + ldCol];
      d[0] = make_uint4(u[0], u[1], u[2], u[3]);
      d[1] = make_uint4(u[4], u[5], u[6], u[7]);
      if (k0 + BKK < K) __builtin_prefetch(s4 + 8, 0, 0);
    }
    // ---- stage B tile (rows = output features, contiguous K) ----
    {
      const float4* s4 = (const float4*)(B + (size_t)(bn + ldRow) * K + k0 + ldCol);
      unsigned u[8];
      #pragma unroll
      for (int c = 0; c < 4; ++c) {
        float4 t = s4[c];
        u[2*c]   = pack2(t.x, t.y);
        u[2*c+1] = pack2(t.z, t.w);
      }
      uint4* d = (uint4*)&Bs[ldRow * LDT + ldCol];
      d[0] = make_uint4(u[0], u[1], u[2], u[3]);
      d[1] = make_uint4(u[4], u[5], u[6], u[7]);
      if (k0 + BKK < K) __builtin_prefetch(s4 + 8, 0, 0);
    }
#if USE_ASYNC_LDS
    if (A_BF16) async_wait0();
#endif
    __syncthreads();

    v16bf af[4], bfr[2];
    #pragma unroll
    for (int mt = 0; mt < 4; ++mt)
      af[mt] = ld_afrag(&As[(wm + mt * 16) * LDT], LDT, 0, lane);
    #pragma unroll
    for (int nt = 0; nt < 2; ++nt)
      bfr[nt] = ld_bfrag(&Bs[(wn + nt * 16) * LDT], LDT, 0, lane);

    #pragma unroll
    for (int mt = 0; mt < 4; ++mt) {
      #pragma unroll
      for (int nt = 0; nt < 2; ++nt)
        acc[mt][nt] = __builtin_amdgcn_wmma_f32_16x16x32_bf16(
            false, af[mt], false, bfr[nt], (short)0, acc[mt][nt], false, false);
    }
    __syncthreads();
  }

  // ---- epilogue ----
  #pragma unroll
  for (int mt = 0; mt < 4; ++mt) {
    #pragma unroll
    for (int nt = 0; nt < 2; ++nt) {
      #pragma unroll
      for (int r = 0; r < 8; ++r) {
        int row = bm + wm + mt * 16 + r + ((lane >> 4) << 3);
        int col = bn + wn + nt * 16 + (lane & 15);
        float v = acc[mt][nt][r];
        if (OUT_BF16) ((unsigned short*)Cvp)[(size_t)row * N + col] = bfbits(v);
        else          ((float*)Cvp)[(size_t)row * N + col] = v;
      }
    }
  }
}

// =======================================================================
// RoPE (in-place on bf16, fuses scale into Q)
// =======================================================================
__global__ __launch_bounds__(256) void rope_kernel(
    unsigned short* __restrict__ x, const float* __restrict__ cosp,
    const float* __restrict__ sinp, int nheads, float scale, int total)
{
  int idx = blockIdx.x * blockDim.x + threadIdx.x;
  if (idx >= total) return;
  int d = idx & 63;
  int h = (idx >> 6) % nheads;
  int s = idx / (nheads * 64);
  unsigned short* p = x + (size_t)s * nheads * 128 + h * 128;
  float x1 = bff(p[d]), x2 = bff(p[d + 64]);
  float c  = cosp[s * 128 + d];
  float sn = sinp[s * 128 + d];
  float o1 = (x1 * c - x2 * sn) * scale;   // rotate_half: (-x2, x1)
  float o2 = (x2 * c + x1 * sn) * scale;
  p[d]      = bfbits(o1);
  p[d + 64] = bfbits(o2);
}

// =======================================================================
// Flash attention: 64 Q rows / block (4 waves), 64-key tiles in LDS,
// softcap + causal + online softmax, all matmuls via v_wmma bf16.
// =======================================================================
#define KSTR 136   // K-tile LDS stride (dims), 272B rows -> 16B aligned
#define VSTR 72    // Vt LDS stride (keys), 144B rows -> 16B aligned
#define PSTR 72    // P LDS stride

__global__ __launch_bounds__(128) void attn_kernel(
    const unsigned short* __restrict__ Q,   // [S, 2048] bf16 (rope+scale)
    const unsigned short* __restrict__ Kc,  // [S, 512]  bf16 (rope)
    const unsigned short* __restrict__ Vc,  // [S, 512]  bf16
    unsigned short* __restrict__ Oc)        // [S, 2048] bf16
{
  const int h    = blockIdx.y;
  const int g    = h >> 2;
  const int mblk = blockIdx.x;
  const int tid  = threadIdx.x;
  const int lane = tid & 31;
  const int wave = tid >> 5;

  __shared__ __align__(16) unsigned short Ks[64 * KSTR];
  __shared__ __align__(16) unsigned short Vt[128 * VSTR];   // transposed [dim][key]
  __shared__ __align__(16) unsigned short Ps[4][16 * PSTR];

  // Q fragments for this wave's 16 rows (4 K-chunks of 32 over D=128)
  v16bf qa[4];
  #pragma unroll
  for (int kk = 0; kk < 4; ++kk)
    qa[kk] = ld_afrag(Q + (size_t)(mblk * 64 + wave * 16) * 2048 + h * 128,
                      2048, kk * 32, lane);

  v8f o[8];
  float mrun[8], lrun[8];
  int rowg[8];
  #pragma unroll
  for (int r = 0; r < 8; ++r) {
    o[r] = zero8();
    mrun[r] = -3.0e38f;
    lrun[r] = 0.0f;
    rowg[r] = mblk * 64 + wave * 16 + r + ((lane >> 4) << 3);
  }

  for (int kb = 0; kb <= mblk; ++kb) {
    __syncthreads();
    // ---- stage K (row-major, async if available) and V (transposed) ----
    {
      const int krow = tid >> 1;
      const unsigned short* kg = Kc + (size_t)(kb * 64 + krow) * 512 + g * 128;
      const unsigned short* vg = Vc + (size_t)(kb * 64 + krow) * 512 + g * 128;
      #pragma unroll
      for (int c = 0; c < 8; ++c) {
        int ch = (tid & 1) * 8 + c;            // 16 chunks of 8 dims
#if USE_ASYNC_LDS
        async_cp16(kg + ch * 8, &Ks[krow * KSTR + ch * 8]);
#else
        uint4 kv = *(const uint4*)(kg + ch * 8);
        *(uint4*)(&Ks[krow * KSTR + ch * 8]) = kv;
#endif
        uint4 vv = *(const uint4*)(vg + ch * 8);
        const unsigned short* e = (const unsigned short*)&vv;
        #pragma unroll
        for (int j = 0; j < 8; ++j)
          Vt[(ch * 8 + j) * VSTR + krow] = e[j];
      }
    }
#if USE_ASYNC_LDS
    async_wait0();
#endif
    __syncthreads();

    // ---- scores: S = Q @ K^T  (16x64 per wave) ----
    v8f sacc[4];
    #pragma unroll
    for (int ns = 0; ns < 4; ++ns) {
      sacc[ns] = zero8();
      #pragma unroll
      for (int kk = 0; kk < 4; ++kk) {
        v16bf kbf = ld_bfrag(&Ks[(ns * 16) * KSTR], KSTR, kk * 32, lane);
        sacc[ns] = __builtin_amdgcn_wmma_f32_16x16x32_bf16(
            false, qa[kk], false, kbf, (short)0, sacc[ns], false, false);
      }
    }

    // ---- softcap + causal + online softmax ----
    #pragma unroll
    for (int r = 0; r < 8; ++r) {
      float sv[4];
      #pragma unroll
      for (int ns = 0; ns < 4; ++ns) {
        int col = kb * 64 + ns * 16 + (lane & 15);
        float s = sacc[ns][r];
        s = 50.0f * tanhf(s * 0.02f);
        if (col > rowg[r]) s = -1.0e9f;
        sv[ns] = s;
      }
      float mx = fmaxf(fmaxf(sv[0], sv[1]), fmaxf(sv[2], sv[3]));
      mx = rmax16(mx);
      float mcur = fmaxf(mrun[r], mx);
      float corr = __expf(mrun[r] - mcur);
      float psum = 0.0f;
      #pragma unroll
      for (int ns = 0; ns < 4; ++ns) {
        float p = __expf(sv[ns] - mcur);
        psum += p;
        Ps[wave][(r + ((lane >> 4) << 3)) * PSTR + ns * 16 + (lane & 15)] = bfbits(p);
      }
      psum = rsum16(psum);
      lrun[r] = lrun[r] * corr + psum;
      mrun[r] = mcur;
      #pragma unroll
      for (int t = 0; t < 8; ++t) o[t][r] *= corr;
    }
    __syncthreads();

    // ---- O += P @ V ----
    v16bf pa[2];
    #pragma unroll
    for (int kc = 0; kc < 2; ++kc)
      pa[kc] = ld_afrag(&Ps[wave][0], PSTR, kc * 32, lane);
    #pragma unroll
    for (int nd = 0; nd < 8; ++nd) {
      #pragma unroll
      for (int kc = 0; kc < 2; ++kc) {
        v16bf vb = ld_bfrag(&Vt[(nd * 16) * VSTR], VSTR, kc * 32, lane);
        o[nd] = __builtin_amdgcn_wmma_f32_16x16x32_bf16(
            false, pa[kc], false, vb, (short)0, o[nd], false, false);
      }
    }
  }

  // ---- normalize + store ----
  #pragma unroll
  for (int nd = 0; nd < 8; ++nd) {
    #pragma unroll
    for (int r = 0; r < 8; ++r) {
      float inv = 1.0f / lrun[r];
      int col = h * 128 + nd * 16 + (lane & 15);
      Oc[(size_t)rowg[r] * 2048 + col] = bfbits(o[nd][r] * inv);
    }
  }
}

// =======================================================================
extern "C" void kernel_launch(void* const* d_in, const int* in_sizes, int n_in,
                              void* d_out, int out_size, void* d_ws, size_t ws_size,
                              hipStream_t stream) {
  const float* hs   = (const float*)d_in[0];   // [4096, 2048]
  const float* cosp = (const float*)d_in[1];   // [1, 4096, 128]
  const float* sinp = (const float*)d_in[2];
  // d_in[3] attention_mask: unused (causal computed analytically)
  const float* wq = (const float*)d_in[4];     // [2048, 2048]
  const float* wk = (const float*)d_in[5];     // [512, 2048]
  const float* wv = (const float*)d_in[6];     // [512, 2048]
  const float* wo = (const float*)d_in[7];     // [2048, 2048]
  float* out = (float*)d_out;

  char* ws = (char*)d_ws;
  unsigned short* qb = (unsigned short*)(ws);                         // 16.78 MB
  unsigned short* kb = (unsigned short*)(ws + 16777216);              //  4.19 MB
  unsigned short* vb = (unsigned short*)(ws + 16777216 + 4194304);    //  4.19 MB
  unsigned short* ab = (unsigned short*)(ws + 16777216 + 8388608);    // 16.78 MB

  dim3 blk(256);
  gemm_bf16_wmma<false, true><<<dim3(16, 32), blk, 0, stream>>>(hs, wq, qb, 4096, 2048, 2048);
  gemm_bf16_wmma<false, true><<<dim3(4, 32),  blk, 0, stream>>>(hs, wk, kb, 4096, 512,  2048);
  gemm_bf16_wmma<false, true><<<dim3(4, 32),  blk, 0, stream>>>(hs, wv, vb, 4096, 512,  2048);

  rope_kernel<<<(4096 * 16 * 64 + 255) / 256, 256, 0, stream>>>(
      qb, cosp, sinp, 16, 0.08838834764831845f, 4096 * 16 * 64);
  rope_kernel<<<(4096 * 4 * 64 + 255) / 256, 256, 0, stream>>>(
      kb, cosp, sinp, 4, 1.0f, 4096 * 4 * 64);

  attn_kernel<<<dim3(64, 16), 128, 0, stream>>>(qb, kb, vb, ab);

  gemm_bf16_wmma<true, false><<<dim3(16, 32), blk, 0, stream>>>(ab, wo, out, 4096, 2048, 2048);
}
